// ScaledMHAttention_68332929679768
// MI455X (gfx1250) — matrix operand order repaired and checked by reading, not compiled, not run on hardware
//
#include <hip/hip_runtime.h>
#include <hip/hip_bf16.h>

typedef unsigned short u16;
typedef __attribute__((ext_vector_type(16))) __bf16 v16bf;
typedef __attribute__((ext_vector_type(8)))  float  v8f;

#define S_LEN  2048
#define DMODEL 2048
#define NHEAD  16
#define HDIM   128

// ---------------------------------------------------------------- CDNA5 async copy (ASYNCcnt path)
// global_load_async_to_lds_b128: VDST = LDS byte address (32-bit VGPR), addr = 64-bit VGPR pair.
__device__ __forceinline__ void async_cp16(const u16* g, u16* l) {
  asm volatile("global_load_async_to_lds_b128 %0, %1, off"
               :: "v"((unsigned)(uintptr_t)l), "v"(g)
               : "memory");
}
#define WAIT_ASYNCCNT(n) asm volatile("s_wait_asynccnt " #n ::: "memory")

// ---------------------------------------------------------------- helpers
__device__ __forceinline__ u16 f2bf(float f) {
  unsigned u = __float_as_uint(f);
  unsigned r = 0x7FFFu + ((u >> 16) & 1u);   // round-to-nearest-even
  return (u16)((u + r) >> 16);
}

union Frag16 { uint4 u[2]; v16bf v; };

// A-matrix fragment (16x32 bf16): lane(m=l%16,h=l/16) holds K = {kb+8h..+7} u {kb+16+8h..+7}
__device__ __forceinline__ v16bf frag_A(const u16* lds, int row, int kbase, int stride, int lh) {
  Frag16 f;
  const u16* p = lds + row * stride + kbase + 8 * lh;
  f.u[0] = *(const uint4*)(p);
  f.u[1] = *(const uint4*)(p + 16);
  return f.v;
}
// B-matrix fragment (32x16 bf16), LDS holds B^T as [n][k]: lane(n=l%16,h) holds K = {kb+8h..+7} u {kb+16+8h..+7}
__device__ __forceinline__ v16bf frag_B(const u16* lds, int row, int kbase, int stride, int lh) {
  Frag16 f;
  const u16* p = lds + row * stride + kbase + 8 * lh;
  f.u[0] = *(const uint4*)(p);
  f.u[1] = *(const uint4*)(p + 16);
  return f.v;
}

__device__ __forceinline__ v8f wmma_bf16(v16bf a, v16bf b, v8f c) {
  return __builtin_amdgcn_wmma_f32_16x16x32_bf16(false, a, false, b, (short)0, c, false, false);
}

__device__ __forceinline__ v8f v8f_zero() {
  v8f z;
#pragma unroll
  for (int i = 0; i < 8; ++i) z[i] = 0.0f;
  return z;
}

// ---------------------------------------------------------------- fp32 -> bf16
__global__ __launch_bounds__(256) void cvt_bf16_kernel(const float* __restrict__ in,
                                                       u16* __restrict__ out, int n) {
  int i = blockIdx.x * 256 + threadIdx.x;
  if (i < n) out[i] = f2bf(in[i]);
}

// ---------------------------------------------------------------- NT GEMM: C[M,N] = A[M,K] * B[N,K]^T
// block tile 128x256, BK=32, double-buffered async LDS staging.
// 8 waves as 2(m) x 4(n); each wave = 64x64 (4x4 WMMA tiles, 16 v8f accumulators).
template <bool F32OUT>
__global__ __launch_bounds__(256) void gemm_nt_kernel(const u16* __restrict__ A,
                                                      const u16* __restrict__ B,
                                                      u16* __restrict__ Cb,
                                                      float* __restrict__ Cf,
                                                      int M, int N, int K) {
  __shared__ alignas(16) u16 asmem[2 * 128 * 40];   // 2 x 10KB
  __shared__ alignas(16) u16 bsmem[2 * 256 * 40];   // 2 x 20KB

  const int tid  = threadIdx.x;
  const int lane = tid & 31;
  const int wave = tid >> 5;
  const int lm   = lane & 15;
  const int lh   = lane >> 4;
  const int bm   = blockIdx.x * 128;
  const int bn   = blockIdx.y * 256;
  const int wm   = (wave & 1) * 64;
  const int wn   = (wave >> 1) * 64;

  v8f acc[4][4];
#pragma unroll
  for (int i = 0; i < 4; ++i)
#pragma unroll
    for (int j = 0; j < 4; ++j) acc[i][j] = v8f_zero();

  // prologue: stage k-chunk 0 into buffer 0 (6 async b128 per thread)
#pragma unroll
  for (int i = tid; i < 512; i += 256) {
    int r = i >> 2, c = (i & 3) << 3;
    async_cp16(&A[(size_t)(bm + r) * K + c], &asmem[r * 40 + c]);
  }
#pragma unroll
  for (int i = tid; i < 1024; i += 256) {
    int r = i >> 2, c = (i & 3) << 3;
    async_cp16(&B[(size_t)(bn + r) * K + c], &bsmem[r * 40 + c]);
  }

  const int nIter = K / 32;
  for (int it = 0; it < nIter; ++it) {
    const int cur = it & 1;
    if (it + 1 < nIter) {
      const int k1 = (it + 1) * 32;
      const int nb = cur ^ 1;
#pragma unroll
      for (int i = tid; i < 512; i += 256) {
        int r = i >> 2, c = (i & 3) << 3;
        async_cp16(&A[(size_t)(bm + r) * K + k1 + c], &asmem[nb * 5120 + r * 40 + c]);
      }
#pragma unroll
      for (int i = tid; i < 1024; i += 256) {
        int r = i >> 2, c = (i & 3) << 3;
        async_cp16(&B[(size_t)(bn + r) * K + k1 + c], &bsmem[nb * 10240 + r * 40 + c]);
      }
      WAIT_ASYNCCNT(6);   // current buffer's 6 done (in-order); next 6 stay in flight
    } else {
      WAIT_ASYNCCNT(0);
    }
    __syncthreads();

    const u16* as_ = &asmem[cur * 5120];
    const u16* bs_ = &bsmem[cur * 10240];
    v16bf af[4], bf[4];
#pragma unroll
    for (int i = 0; i < 4; ++i) af[i] = frag_A(as_, wm + i * 16 + lm, 0, 40, lh);
#pragma unroll
    for (int j = 0; j < 4; ++j) bf[j] = frag_B(bs_, wn + j * 16 + lm, 0, 40, lh);
#pragma unroll
    for (int i = 0; i < 4; ++i)
#pragma unroll
      for (int j = 0; j < 4; ++j) acc[i][j] = wmma_bf16(af[i], bf[j], acc[i][j]);
    __syncthreads();   // done reading 'cur' before it is overwritten next iteration
  }

  // epilogue: C tile layout -> VGPR v holds rows v (lanes 0-15) / v+8 (lanes 16-31), col = lane%16
#pragma unroll
  for (int i = 0; i < 4; ++i)
#pragma unroll
    for (int j = 0; j < 4; ++j)
#pragma unroll
      for (int v = 0; v < 8; ++v) {
        int row = bm + wm + i * 16 + v + 8 * lh;
        int col = bn + wn + j * 16 + lm;
        if (F32OUT) Cf[(size_t)row * N + col] = acc[i][j][v];
        else        Cb[(size_t)row * N + col] = f2bf(acc[i][j][v]);
      }
}

// ---------------------------------------------------------------- flash attention (softmax1)
// block = (q-block of 128, head); 8 waves, wave owns 16 q-rows.
__global__ __launch_bounds__(256) void attn_fa_kernel(const u16* __restrict__ Qh,
                                                      const u16* __restrict__ Kh,
                                                      const u16* __restrict__ Vh,
                                                      const float* __restrict__ scalars,
                                                      u16* __restrict__ Oattn) {
  extern __shared__ u16 smem[];
  const int ST = 136;                 // padded row stride (keeps 16B alignment)
  u16* qs = smem;                     // [128][ST] Q block (row-major over HD)
  u16* ks = qs + 128 * ST;            // [128][ST] K block rows over HD
  u16* vt = ks + 128 * ST;            // [128][ST] V block TRANSPOSED: [d][k]
  u16* ps = vt + 128 * ST;            // [128][ST] probabilities (bf16)

  const int tid  = threadIdx.x;
  const int lane = tid & 31;
  const int w    = tid >> 5;
  const int lm   = lane & 15;
  const int lh   = lane >> 4;
  const int qb   = blockIdx.x * 128;
  const int h    = blockIdx.y;
  const int cb   = h * HDIM;
  const float qk_scale = 0.08838834764831845f;  // 1/sqrt(128)

  // stage Q block asynchronously (128 rows x 128 cols bf16)
#pragma unroll
  for (int i = tid; i < 2048; i += 256) {
    int r = i >> 4, c = (i & 15) << 3;
    async_cp16(&Qh[(size_t)(qb + r) * DMODEL + cb + c], &qs[r * ST + c]);
  }
  WAIT_ASYNCCNT(0);
  __syncthreads();

  v16bf qf[4];
#pragma unroll
  for (int kk = 0; kk < 4; ++kk) qf[kk] = frag_A(qs, w * 16 + lm, kk * 32, ST, lh);

  // online softmax1 state: virtual 0-logit  ->  m=0, l=exp(0-0)=1
  float mrow[8], lrow[8];
  v8f oacc[8];
#pragma unroll
  for (int v = 0; v < 8; ++v) { mrow[v] = 0.0f; lrow[v] = 1.0f; }
#pragma unroll
  for (int t = 0; t < 8; ++t) oacc[t] = v8f_zero();

  for (int kb = 0; kb < S_LEN; kb += 128) {
    __syncthreads();  // previous iteration done with ks/vt
    // stage K rows (async) and V^T (transpose during staging, VGPR round-trip)
#pragma unroll
    for (int i = tid; i < 2048; i += 256) {
      int r = i >> 4, c = (i & 15) << 3;
      async_cp16(&Kh[(size_t)(kb + r) * DMODEL + cb + c], &ks[r * ST + c]);
      uint4 vv = *(const uint4*)&Vh[(size_t)(kb + r) * DMODEL + cb + c];
      const u16* pv = (const u16*)&vv;
#pragma unroll
      for (int j = 0; j < 8; ++j) vt[(c + j) * ST + r] = pv[j];
    }
    WAIT_ASYNCCNT(0);
    __syncthreads();

    // S = (Q * scale) K^T : 8 n-tiles x 4 k-chunks
    v8f st[8];
#pragma unroll
    for (int t = 0; t < 8; ++t) {
      st[t] = v8f_zero();
#pragma unroll
      for (int kk = 0; kk < 4; ++kk) {
        v16bf kf = frag_B(ks, t * 16 + lm, kk * 32, ST, lh);
        st[t] = wmma_bf16(qf[kk], kf, st[t]);
      }
    }
#pragma unroll
    for (int t = 0; t < 8; ++t)
#pragma unroll
      for (int v = 0; v < 8; ++v) st[t][v] *= qk_scale;

    // row max across 128 columns: per-VGPR over tiles, then 16-lane butterfly
    float rmax[8];
#pragma unroll
    for (int v = 0; v < 8; ++v) {
      float r = st[0][v];
#pragma unroll
      for (int t = 1; t < 8; ++t) r = fmaxf(r, st[t][v]);
#pragma unroll
      for (int d = 1; d < 16; d <<= 1) r = fmaxf(r, __shfl_xor(r, d, 32));
      rmax[v] = r;
    }
    float alpha[8];
#pragma unroll
    for (int v = 0; v < 8; ++v) {
      float mn = fmaxf(mrow[v], rmax[v]);
      alpha[v] = __expf(mrow[v] - mn);
      mrow[v]  = mn;
    }
    // P = exp(S - m), row sums
    float rsum[8];
#pragma unroll
    for (int v = 0; v < 8; ++v) rsum[v] = 0.0f;
#pragma unroll
    for (int t = 0; t < 8; ++t)
#pragma unroll
      for (int v = 0; v < 8; ++v) {
        float p = __expf(st[t][v] - mrow[v]);
        st[t][v] = p;
        rsum[v] += p;
      }
#pragma unroll
    for (int v = 0; v < 8; ++v) {
      float r = rsum[v];
#pragma unroll
      for (int d = 1; d < 16; d <<= 1) r += __shfl_xor(r, d, 32);
      lrow[v] = lrow[v] * alpha[v] + r;
    }
    // rescale O, spill P to LDS (wave-private rows; per-wave DS ordering covers the readback)
#pragma unroll
    for (int t = 0; t < 8; ++t)
#pragma unroll
      for (int v = 0; v < 8; ++v) {
        oacc[t][v] *= alpha[v];
        ps[(w * 16 + v + 8 * lh) * ST + (t * 16 + lm)] = f2bf(st[t][v]);
      }
    // O += P * V  (K-dim = 128 k-rows; B fragments from transposed V)
    v16bf pf[4];
#pragma unroll
    for (int kk = 0; kk < 4; ++kk) pf[kk] = frag_A(ps, w * 16 + lm, kk * 32, ST, lh);
#pragma unroll
    for (int t = 0; t < 8; ++t)
#pragma unroll
      for (int kk = 0; kk < 4; ++kk) {
        v16bf vf = frag_B(vt, t * 16 + lm, kk * 32, ST, lh);
        oacc[t] = wmma_bf16(pf[kk], vf, oacc[t]);
      }
  }

  // epilogue: attn * scalars[h] / denom  (denom already includes the softmax1 "+exp(-m)")
  const float sc = scalars[h];
#pragma unroll
  for (int t = 0; t < 8; ++t)
#pragma unroll
    for (int v = 0; v < 8; ++v) {
      int row = qb + w * 16 + v + 8 * lh;
      int col = cb + t * 16 + lm;
      Oattn[(size_t)row * DMODEL + col] = f2bf(oacc[t][v] * sc / lrow[v]);
    }
}

// ---------------------------------------------------------------- host
extern "C" void kernel_launch(void* const* d_in, const int* in_sizes, int n_in,
                              void* d_out, int out_size, void* d_ws, size_t ws_size,
                              hipStream_t stream) {
  (void)in_sizes; (void)n_in; (void)out_size; (void)ws_size;
  const float* query   = (const float*)d_in[0];
  const float* key_    = (const float*)d_in[1];
  const float* value   = (const float*)d_in[2];
  const float* Wq      = (const float*)d_in[3];
  const float* Wk      = (const float*)d_in[4];
  const float* Wv      = (const float*)d_in[5];
  const float* Wo      = (const float*)d_in[6];
  const float* scalars = (const float*)d_in[7];

  const size_t NE = (size_t)S_LEN * DMODEL;  // 4M elements
  u16* qB   = (u16*)d_ws;
  u16* kB   = qB  + NE;
  u16* vB   = kB  + NE;
  u16* WqB  = vB  + NE;
  u16* WkB  = WqB + NE;
  u16* WvB  = WkB + NE;
  u16* WoB  = WvB + NE;
  u16* Qh   = WoB + NE;
  u16* Kh   = Qh  + NE;
  u16* Vh   = Kh  + NE;
  u16* attn = Vh  + NE;   // total 11 * 8MB = 88MB bf16 workspace

  const int cvtBlocks = (int)(NE / 256);
  cvt_bf16_kernel<<<cvtBlocks, 256, 0, stream>>>(query, qB,  (int)NE);
  cvt_bf16_kernel<<<cvtBlocks, 256, 0, stream>>>(key_,  kB,  (int)NE);
  cvt_bf16_kernel<<<cvtBlocks, 256, 0, stream>>>(value, vB,  (int)NE);
  cvt_bf16_kernel<<<cvtBlocks, 256, 0, stream>>>(Wq,    WqB, (int)NE);
  cvt_bf16_kernel<<<cvtBlocks, 256, 0, stream>>>(Wk,    WkB, (int)NE);
  cvt_bf16_kernel<<<cvtBlocks, 256, 0, stream>>>(Wv,    WvB, (int)NE);
  cvt_bf16_kernel<<<cvtBlocks, 256, 0, stream>>>(Wo,    WoB, (int)NE);

  dim3 ggrid(S_LEN / 128, DMODEL / 256);
  gemm_nt_kernel<false><<<ggrid, 256, 0, stream>>>(qB, WqB, Qh, nullptr, S_LEN, DMODEL, DMODEL);
  gemm_nt_kernel<false><<<ggrid, 256, 0, stream>>>(kB, WkB, Kh, nullptr, S_LEN, DMODEL, DMODEL);
  gemm_nt_kernel<false><<<ggrid, 256, 0, stream>>>(vB, WvB, Vh, nullptr, S_LEN, DMODEL, DMODEL);

  const int shmem = 4 * 128 * 136 * (int)sizeof(u16);  // 136 KB dynamic LDS
  hipFuncSetAttribute((const void*)attn_fa_kernel,
                      hipFuncAttributeMaxDynamicSharedMemorySize, shmem);
  dim3 agrid(S_LEN / 128, NHEAD);
  attn_fa_kernel<<<agrid, 256, shmem, stream>>>(Qh, Kh, Vh, scalars, attn);

  gemm_nt_kernel<true><<<ggrid, 256, 0, stream>>>(attn, WoB, nullptr, (float*)d_out,
                                                  S_LEN, DMODEL, DMODEL);
}
